// ZINBDecoder_30889404793298
// MI455X (gfx1250) — compile-verified, matrix operand order checked by reading, not run on hardware
//
#include <hip/hip_runtime.h>
#include <hip/hip_bf16.h>

// ZINB decoder edge kernel for MI455X (gfx1250, wave32).
// Per 16-edge tile (one wave): build A = (c[src] ⊙ g[dst]) as 16x64 f16,
// B = [Wm|Wd|Wp|0...] as 64x16 f16, D = A@B via two v_wmma_f32_16x16x32_f16,
// then ZINB activation epilogue on lanes 0..15.

typedef __attribute__((ext_vector_type(16))) _Float16 v16h;  // WMMA operand type
typedef __attribute__((ext_vector_type(2)))  __fp16   v2h;   // cvt_pkrtz result type
typedef __attribute__((ext_vector_type(8)))  float    v8f;

#define D_FEAT 64

union HFrag { v2h h[8]; v16h v; };   // same bits, two clang types

__device__ __forceinline__ v2h pk2(float a, float b) {
    // single v_cvt_pk_rtz_f16_f32
    return __builtin_amdgcn_cvt_pkrtz(a, b);
}

__global__ __launch_bounds__(256) void zinb_edge_wmma(
    const float* __restrict__ c_feat, const float* __restrict__ g_feat,
    const float* __restrict__ cs_factor, const float* __restrict__ gs_factor,
    const int* __restrict__ src, const int* __restrict__ dst,
    const float* __restrict__ Wm, const float* __restrict__ bm,
    const float* __restrict__ Wd, const float* __restrict__ bd,
    const float* __restrict__ Wp, const float* __restrict__ bp,
    float* __restrict__ out, int E)
{
    __shared__ float lds[8][16 * 16];            // one 16x16 f32 D-tile per wave
    const int lane = threadIdx.x & 31;
    const int wave = threadIdx.x >> 5;
    const int half = lane >> 4;                  // 0: K chunks {0-7,16-23}, 1: {8-15,24-31}
    const int m    = lane & 15;                  // row (edge-in-tile) for A; col (head) for B
    const int kb   = half * 8;

    // ---- B fragments: columns 0..2 = Wm,Wd,Wp; columns 3..15 = 0 (loop invariant) ----
    HFrag B1, B2;
    {
        const float* Wh = (m == 0) ? Wm : (m == 1) ? Wd : (m == 2) ? Wp : Wm;
        const bool live = (m < 3);
        #pragma unroll
        for (int v = 0; v < 4; ++v) {
            B1.h[v]     = live ? pk2(Wh[kb + 2*v],      Wh[kb + 2*v + 1])      : pk2(0.f, 0.f);
            B1.h[4 + v] = live ? pk2(Wh[16 + kb + 2*v], Wh[16 + kb + 2*v + 1]) : pk2(0.f, 0.f);
            B2.h[v]     = live ? pk2(Wh[32 + kb + 2*v], Wh[32 + kb + 2*v + 1]) : pk2(0.f, 0.f);
            B2.h[4 + v] = live ? pk2(Wh[48 + kb + 2*v], Wh[48 + kb + 2*v + 1]) : pk2(0.f, 0.f);
        }
    }
    const float bmv = bm[0], bdv = bd[0], bpv = bp[0];

    const int nTiles = E >> 4;                   // E = 4M -> 250k tiles of 16 edges
    const int waveId = blockIdx.x * (blockDim.x >> 5) + wave;
    const int nWaves = gridDim.x * (blockDim.x >> 5);

    for (int t = waveId; t < nTiles; t += nWaves) {
        const int base = t << 4;

        // stream-ahead prefetch of the edge index arrays (global_prefetch_b8)
        const int pf = (base + 4096 < E) ? base + 4096 : base;
        __builtin_prefetch(src + pf, 0, 3);
        __builtin_prefetch(dst + pf, 0, 3);

        const int s = src[base + m];
        const int j = dst[base + m];
        const float* crow = c_feat + (long)s * D_FEAT;   // 2.56 MB table, L2-resident
        const float* grow = g_feat + (long)j * D_FEAT;   // 0.5 MB table, L2-resident

        // ---- A fragments: h = c ⊙ g, converted f32 -> packed f16 ----
        HFrag A1, A2;
        #pragma unroll
        for (int blk = 0; blk < 2; ++blk) {
            const int k0 = blk * 32 + kb;        // K = k0 .. k0+7
            const int k1 = blk * 32 + 16 + kb;   // K = k1 .. k1+7
            float4 ca = *(const float4*)(crow + k0);
            float4 cb = *(const float4*)(crow + k0 + 4);
            float4 cc = *(const float4*)(crow + k1);
            float4 cd = *(const float4*)(crow + k1 + 4);
            float4 ga = *(const float4*)(grow + k0);
            float4 gb = *(const float4*)(grow + k0 + 4);
            float4 gc = *(const float4*)(grow + k1);
            float4 gd = *(const float4*)(grow + k1 + 4);
            HFrag& A = blk ? A2 : A1;
            A.h[0] = pk2(ca.x * ga.x, ca.y * ga.y);
            A.h[1] = pk2(ca.z * ga.z, ca.w * ga.w);
            A.h[2] = pk2(cb.x * gb.x, cb.y * gb.y);
            A.h[3] = pk2(cb.z * gb.z, cb.w * gb.w);
            A.h[4] = pk2(cc.x * gc.x, cc.y * gc.y);
            A.h[5] = pk2(cc.z * gc.z, cc.w * gc.w);
            A.h[6] = pk2(cd.x * gd.x, cd.y * gd.y);
            A.h[7] = pk2(cd.z * gd.z, cd.w * gd.w);
        }

        // ---- D = A @ B : two 16x16x32 f16 WMMAs cover K=64 ----
        v8f acc = {};
        acc = __builtin_amdgcn_wmma_f32_16x16x32_f16(false, A1.v, false, B1.v,
                                                     (short)0, acc, false, false);
        acc = __builtin_amdgcn_wmma_f32_16x16x32_f16(false, A2.v, false, B2.v,
                                                     (short)0, acc, false, false);

        // scatter D to wave-private LDS: D[M][N], M = r + 8*half, N = m
        #pragma unroll
        for (int r = 0; r < 8; ++r)
            lds[wave][(r + 8 * half) * 16 + m] = acc[r];

        // ---- ZINB epilogue: lanes 0..15, one edge each (in-order DS, same wave) ----
        if (half == 0) {
            const float dm = lds[wave][m * 16 + 0];   // h . Wm
            const float dd = lds[wave][m * 16 + 1];   // h . Wd
            const float dp = lds[wave][m * 16 + 2];   // h . Wp
            const float cs = cs_factor[s];
            const float gs = gs_factor[j];

            const float mu_  = 1.f / (1.f + __expf(-(dm + bmv)));        // sigmoid
            const float pi   = 1.f / (1.f + __expf(-(dp + bpv)));        // sigmoid
            const float x    = gs * (dd + bdv);
            const float sp   = (x > 20.f) ? x : __logf(1.f + __expf(x)); // softplus
            const float disp = fminf(fmaxf(sp, 1e-4f), 1e4f);
            const float mu   = cs * fminf(fmaxf(__expf(gs * mu_) - 1.f, 1e-5f), 1e6f);

            const int e = base + m;
            out[e]         = mu;
            out[E + e]     = disp;
            out[2 * E + e] = pi;
        }
    }
}

extern "C" void kernel_launch(void* const* d_in, const int* in_sizes, int n_in,
                              void* d_out, int out_size, void* d_ws, size_t ws_size,
                              hipStream_t stream) {
    const float* c_feat = (const float*)d_in[0];
    const float* g_feat = (const float*)d_in[1];
    const float* cs     = (const float*)d_in[2];
    const float* gs     = (const float*)d_in[3];
    const int*   src    = (const int*)  d_in[4];
    const int*   dst    = (const int*)  d_in[5];
    const float* Wm     = (const float*)d_in[6];
    const float* bm     = (const float*)d_in[7];
    const float* Wd     = (const float*)d_in[8];
    const float* bd     = (const float*)d_in[9];
    const float* Wp     = (const float*)d_in[10];
    const float* bp     = (const float*)d_in[11];

    const int E = in_sizes[4];          // 4,000,000 edges (multiple of 16)

    // 1024 blocks x 256 threads = 8192 wave32s grid-striding over 250k edge tiles
    zinb_edge_wmma<<<1024, 256, 0, stream>>>(c_feat, g_feat, cs, gs, src, dst,
                                             Wm, bm, Wd, bd, Wp, bp,
                                             (float*)d_out, E);
}
